// MiniAttentionBlock_57019985822328
// MI455X (gfx1250) — matrix operand order repaired and checked by reading, not compiled, not run on hardware
//
#include <hip/hip_runtime.h>
#include <hip/hip_bf16.h>

// ---------------------------------------------------------------------------
// Types
// ---------------------------------------------------------------------------
typedef __bf16 bf16;
typedef __attribute__((ext_vector_type(16))) __bf16 v16bf;
typedef __attribute__((ext_vector_type(8)))  float  v8f;
typedef __attribute__((ext_vector_type(4)))  unsigned int u32x4;
typedef __attribute__((ext_vector_type(4)))  __bf16 v4bf;
typedef __attribute__((ext_vector_type(8)))  int i32x8;
typedef __attribute__((ext_vector_type(4)))  int i32x4;

union AFrag { v16bf v; u32x4 q[2]; };   // 16x32 bf16 A fragment (8 VGPRs/lane)
union BFrag { v16bf v; u32x4 q[2]; };   // 32x16 bf16 B fragment

constexpr int H   = 1024;
constexpr int T   = 2048;
constexpr int NB  = 4;     // batch
constexpr int NH  = 8;     // heads
constexpr int HD  = 128;   // head dim
constexpr int LDT = 40;    // padded LDS stride (elements) for 32-wide tiles
constexpr float ATT_SCALE = 0.08838834764831845f;  // 1/sqrt(128)

#if defined(__has_builtin)
#if __has_builtin(__builtin_amdgcn_tensor_load_to_lds)
#define HAVE_TDM 1
#endif
#endif

__device__ __forceinline__ bf16 f2bf(float f) {
  union { float f; unsigned u; } a; a.f = f;
  unsigned r = (a.u + 0x7FFFu + ((a.u >> 16) & 1u)) >> 16;
  union { unsigned short s; bf16 b; } o; o.s = (unsigned short)r;
  return o.b;
}

__device__ __forceinline__ v8f wmma_bf16(v16bf a, v16bf b, v8f c) {
  // D(16x16,f32) = A(16x32,bf16) x B(32x16,bf16) + C
  return __builtin_amdgcn_wmma_f32_16x16x32_bf16(false, a, false, b,
                                                 (short)0, c, false, false);
}

// ---------------------------------------------------------------------------
// Tensor Data Mover: 2D tile (tile_w x tile_h, 2-byte elems) -> padded LDS.
// D# layout per CDNA5 ISA. pad_interval: 2^(v+1) DWORDs
// between pads; pad_amount: (v+1) DWORDs of pad. Issue from one wave only.
// ---------------------------------------------------------------------------
__device__ __forceinline__ void tdm_load_2d(const void* gsrc, unsigned lds_off,
                                            unsigned tile_w, unsigned tile_h,
                                            unsigned row_stride,
                                            unsigned pad_interval,
                                            unsigned pad_amount) {
#ifdef HAVE_TDM
  unsigned long long ga = (unsigned long long)(uintptr_t)gsrc;
  u32x4 g0;
  g0[0] = 1u;                                    // count=1 (valid user D#)
  g0[1] = lds_off;                               // lds_addr (bytes)
  g0[2] = (unsigned)(ga & 0xffffffffu);          // global_addr[31:0]
  g0[3] = (unsigned)((ga >> 32) & 0x01ffffffu)   // global_addr[56:32]
          | 0x80000000u;                         // type = 2 ("image")
  i32x8 g1;
  g1[0] = (int)((1u << 16)                       // data_size = 2 bytes
                | (1u << 20)                     // pad_enable
                | (pad_interval << 22) | (pad_amount << 25));
  g1[1] = (int)((row_stride & 0xffffu) << 16);   // tensor_dim0[15:0]
  g1[2] = (int)((row_stride >> 16) & 0xffffu);   // tensor_dim0[31:16]
  g1[3] = (int)(16u | (tile_w << 16));           // tensor_dim1=1<<20; tile_dim0
  g1[4] = (int)(tile_h & 0xffffu);               // tile_dim1 (tile_dim2=0)
  g1[5] = (int)row_stride;                       // tensor_dim0_stride[31:0]
  g1[6] = 0;                                     // stride[47:32], dim1_stride
  g1[7] = 0;
  i32x4 gz = {0, 0, 0, 0};
#if __clang_major__ >= 23
  i32x8 gz8 = {0, 0, 0, 0, 0, 0, 0, 0};
  __builtin_amdgcn_tensor_load_to_lds(g0, g1, gz, gz, gz8, 0);
#else
  __builtin_amdgcn_tensor_load_to_lds(g0, g1, gz, gz, 0);
#endif
#else
  (void)gsrc; (void)lds_off; (void)tile_w; (void)tile_h;
  (void)row_stride; (void)pad_interval; (void)pad_amount;
#endif
}

// ---------------------------------------------------------------------------
// 1) fp32 -> bf16 weight conversion (4 matrices, selected by blockIdx.y)
// ---------------------------------------------------------------------------
__global__ __launch_bounds__(256) void cvt_w_kernel(
    const float* __restrict__ s0, const float* __restrict__ s1,
    const float* __restrict__ s2, const float* __restrict__ s3,
    bf16* __restrict__ d0, bf16* __restrict__ d1,
    bf16* __restrict__ d2, bf16* __restrict__ d3) {
  const float* s = s0; bf16* d = d0;
  if      (blockIdx.y == 1) { s = s1; d = d1; }
  else if (blockIdx.y == 2) { s = s2; d = d2; }
  else if (blockIdx.y == 3) { s = s3; d = d3; }
  int i = blockIdx.x * 256 + threadIdx.x;
  d[i] = f2bf(s[i]);
}

// ---------------------------------------------------------------------------
// 2) LayerNorm (one block per token row of 1024) -> bf16
// ---------------------------------------------------------------------------
__global__ __launch_bounds__(256) void ln_kernel(
    const float* __restrict__ x, const float* __restrict__ gamma,
    const float* __restrict__ beta, bf16* __restrict__ xn) {
  int row = blockIdx.x;
  int t   = threadIdx.x;
  const float4* xr = (const float4*)(x + (size_t)row * H);
  float4 v = xr[t];
  float s1 = v.x + v.y + v.z + v.w;
  float s2 = v.x * v.x + v.y * v.y + v.z * v.z + v.w * v.w;
#pragma unroll
  for (int m = 16; m >= 1; m >>= 1) {
    s1 += __shfl_xor(s1, m, 32);
    s2 += __shfl_xor(s2, m, 32);
  }
  __shared__ float r1[8], r2[8];
  int wid = t >> 5, lane = t & 31;
  if (lane == 0) { r1[wid] = s1; r2[wid] = s2; }
  __syncthreads();
  s1 = 0.f; s2 = 0.f;
#pragma unroll
  for (int i = 0; i < 8; ++i) { s1 += r1[i]; s2 += r2[i]; }
  float mean = s1 * (1.0f / H);
  float var  = s2 * (1.0f / H) - mean * mean;
  float rstd = rsqrtf(var + 1e-5f);
  float4 gv = ((const float4*)gamma)[t];
  float4 bv = ((const float4*)beta)[t];
  union { v4bf v; bf16 e[4]; } o;
  o.e[0] = f2bf((v.x - mean) * rstd * gv.x + bv.x);
  o.e[1] = f2bf((v.y - mean) * rstd * gv.y + bv.y);
  o.e[2] = f2bf((v.z - mean) * rstd * gv.z + bv.z);
  o.e[3] = f2bf((v.w - mean) * rstd * gv.w + bv.w);
  *(v4bf*)(xn + (size_t)row * H + t * 4) = o.v;
}

// ---------------------------------------------------------------------------
// WMMA GEMM mainloop: C(128x64 block tile) = A(MxK) * W(NxK)^T
// 256 threads = 8 waves in a 4(M) x 2(N) grid; each wave: 32x32 via 2x2 frags
// A/W tiles staged into padded LDS by the Tensor Data Mover (wave 0 issues).
// ---------------------------------------------------------------------------
__device__ __forceinline__ void gemm_mainloop(
    const bf16* __restrict__ A, const bf16* __restrict__ W,
    int m0, int n0, v8f c[2][2]) {
  __shared__ bf16 lA[128 * LDT];
  __shared__ bf16 lB[64 * LDT];
  const int t    = threadIdx.x;
  const int lane = t & 31, wid = t >> 5;
  const int wm   = wid & 3, wn = wid >> 2;
  const int l15  = lane & 15, lh = lane >> 4;

#pragma unroll
  for (int i = 0; i < 2; ++i)
#pragma unroll
    for (int j = 0; j < 2; ++j)
#pragma unroll
      for (int e = 0; e < 8; ++e) c[i][j][e] = 0.0f;

  for (int kk = 0; kk < H; kk += 32) {
#ifdef HAVE_TDM
    if (wid == 0) {
      // 32-elem rows (16 DWORDs) + 4 DWORDs pad -> LDS stride 40 elems (LDT)
      tdm_load_2d(A + (size_t)m0 * H + kk, (unsigned)(uintptr_t)lA,
                  32, 128, H, /*pad_interval=16dw*/3, /*pad=4dw*/3);
      tdm_load_2d(W + (size_t)n0 * H + kk, (unsigned)(uintptr_t)lB,
                  32, 64, H, 3, 3);
      __builtin_amdgcn_s_wait_tensorcnt(0);
    }
#else
    {
      int r = t >> 1, hc = (t & 1) * 16;
      const u32x4* g = (const u32x4*)(A + (size_t)(m0 + r) * H + kk + hc);
      u32x4 a0 = g[0], a1 = g[1];
      u32x4* s = (u32x4*)(lA + r * LDT + hc);
      s[0] = a0; s[1] = a1;
    }
    {
      int r = t >> 2, q = (t & 3) * 8;
      const u32x4* g = (const u32x4*)(W + (size_t)(n0 + r) * H + kk + q);
      *(u32x4*)(lB + r * LDT + q) = g[0];
    }
#endif
    __syncthreads();
    if (kk + 32 < H) {  // hint next K-slab into cache (global_prefetch_b8)
      __builtin_prefetch(A + (size_t)(m0 + (t >> 1)) * H + kk + 32, 0, 1);
      __builtin_prefetch(W + (size_t)(n0 + (t >> 2)) * H + kk + 32, 0, 1);
    }
    AFrag a[2]; BFrag b[2];
#pragma unroll
    for (int i = 0; i < 2; ++i) {
      int m = wm * 32 + i * 16 + l15;
      a[i].q[0] = *(const u32x4*)(lA + m * LDT + lh * 8);
      a[i].q[1] = *(const u32x4*)(lA + m * LDT + lh * 8 + 16);
    }
#pragma unroll
    for (int j = 0; j < 2; ++j) {
      int n = wn * 32 + j * 16 + l15;
      b[j].q[0] = *(const u32x4*)(lB + n * LDT + lh * 8);
      b[j].q[1] = *(const u32x4*)(lB + n * LDT + lh * 8 + 16);
    }
#pragma unroll
    for (int i = 0; i < 2; ++i)
#pragma unroll
      for (int j = 0; j < 2; ++j)
        c[i][j] = wmma_bf16(a[i].v, b[j].v, c[i][j]);
    __syncthreads();
  }
}

// 3) Q/K/V projections: Y = xn @ W^T (bf16 out). blockIdx.z selects matrix.
__global__ __launch_bounds__(256) void gemm_qkv_kernel(
    const bf16* __restrict__ A,
    const bf16* __restrict__ W0, const bf16* __restrict__ W1,
    const bf16* __restrict__ W2,
    bf16* __restrict__ O0, bf16* __restrict__ O1, bf16* __restrict__ O2) {
  int m0 = blockIdx.y * 128, n0 = blockIdx.x * 64;
  const bf16* W = W0; bf16* O = O0;
  if      (blockIdx.z == 1) { W = W1; O = O1; }
  else if (blockIdx.z == 2) { W = W2; O = O2; }
  v8f c[2][2];
  gemm_mainloop(A, W, m0, n0, c);
  const int lane = threadIdx.x & 31, wid = threadIdx.x >> 5;
  const int wm = wid & 3, wn = wid >> 2, l15 = lane & 15, lh = lane >> 4;
#pragma unroll
  for (int i = 0; i < 2; ++i)
#pragma unroll
    for (int j = 0; j < 2; ++j)
#pragma unroll
      for (int r = 0; r < 8; ++r) {
        int m = m0 + wm * 32 + i * 16 + r + 8 * lh;
        int n = n0 + wn * 32 + j * 16 + l15;
        O[(size_t)m * H + n] = f2bf(c[i][j][r]);
      }
}

// 5) Output projection + residual: out = attnO @ Wo^T + x (fp32 out)
__global__ __launch_bounds__(256) void gemm_resid_kernel(
    const bf16* __restrict__ A, const bf16* __restrict__ W,
    const float* __restrict__ resid, float* __restrict__ out) {
  int m0 = blockIdx.y * 128, n0 = blockIdx.x * 64;
  v8f c[2][2];
  gemm_mainloop(A, W, m0, n0, c);
  const int lane = threadIdx.x & 31, wid = threadIdx.x >> 5;
  const int wm = wid & 3, wn = wid >> 2, l15 = lane & 15, lh = lane >> 4;
#pragma unroll
  for (int i = 0; i < 2; ++i)
#pragma unroll
    for (int j = 0; j < 2; ++j)
#pragma unroll
      for (int r = 0; r < 8; ++r) {
        int m = m0 + wm * 32 + i * 16 + r + 8 * lh;
        int n = n0 + wn * 32 + j * 16 + l15;
        out[(size_t)m * H + n] = c[i][j][r] + resid[(size_t)m * H + n];
      }
}

// ---------------------------------------------------------------------------
// 4) Causal flash-attention: block = 64 query rows of one (batch, head);
//    4 waves, each owns 16 query rows. Key tiles of 32, online softmax.
//    K tile staged by TDM (128-elem rows, 64dw + 4dw pad -> stride 136 elems).
// ---------------------------------------------------------------------------
__global__ __launch_bounds__(128) void attn_kernel(
    const bf16* __restrict__ Qb, const bf16* __restrict__ Kb,
    const bf16* __restrict__ Vb, bf16* __restrict__ Ob) {
  __shared__ bf16 sK [32 * 136];      // K tile: [key][dim], padded
  __shared__ bf16 sVt[128 * LDT];     // V tile transposed: [dim][key]
  __shared__ bf16 sP [4 * 16 * LDT];  // per-wave P relayout scratch

  const int b  = blockIdx.z, hh = blockIdx.y;
  const int q0 = blockIdx.x * 64;
  const int t  = threadIdx.x, lane = t & 31, w = t >> 5;
  const int l15 = lane & 15, lh = lane >> 4;
  const int qw  = q0 + w * 16;
  const size_t base = ((size_t)b * T) * H + hh * HD;

  // Q fragments for this wave's 16 rows (A layout, d = 4 chunks of 32)
  AFrag qf[4];
#pragma unroll
  for (int i = 0; i < 4; ++i) {
    const bf16* p = Qb + base + (size_t)(qw + l15) * H + i * 32 + lh * 8;
    qf[i].q[0] = *(const u32x4*)p;
    qf[i].q[1] = *(const u32x4*)(p + 16);
  }

  v8f o[8];
#pragma unroll
  for (int c = 0; c < 8; ++c)
#pragma unroll
    for (int e = 0; e < 8; ++e) o[c][e] = 0.0f;
  float mst[8], lst[8];
#pragma unroll
  for (int r = 0; r < 8; ++r) { mst[r] = -3.0e38f; lst[r] = 0.0f; }

  const int ntiles = (q0 + 64) / 32;  // uniform across the block (causal)
  for (int tile = 0; tile < ntiles; ++tile) {
    const int j0 = tile * 32;
    __syncthreads();
#ifdef HAVE_TDM
    if (w == 0)
      tdm_load_2d(Kb + base + (size_t)j0 * H, (unsigned)(uintptr_t)sK,
                  128, 32, H, /*pad_interval=64dw*/5, /*pad=4dw*/3);
#else
    {
      int key = t >> 2, ch = (t & 3) * 32;
      const u32x4* g = (const u32x4*)(Kb + base + (size_t)(j0 + key) * H + ch);
      u32x4* s = (u32x4*)(sK + key * 136 + ch);
      s[0] = g[0]; s[1] = g[1]; s[2] = g[2]; s[3] = g[3];
    }
#endif
    // V tile -> LDS transposed [dim][key] (manual transpose; TDM can't)
    {
      int key = t >> 2, ch = (t & 3) * 32;
      union { u32x4 q[4]; bf16 e[32]; } vt;
      const u32x4* g = (const u32x4*)(Vb + base + (size_t)(j0 + key) * H + ch);
      vt.q[0] = g[0]; vt.q[1] = g[1]; vt.q[2] = g[2]; vt.q[3] = g[3];
#pragma unroll
      for (int u = 0; u < 32; ++u) sVt[(ch + u) * LDT + key] = vt.e[u];
    }
#ifdef HAVE_TDM
    if (w == 0) __builtin_amdgcn_s_wait_tensorcnt(0);
#endif
    __syncthreads();

    // S = Q K^T for 32 keys (two 16-col halves)
    v8f s0, s1;
#pragma unroll
    for (int e = 0; e < 8; ++e) { s0[e] = 0.0f; s1[e] = 0.0f; }
#pragma unroll
    for (int i = 0; i < 4; ++i) {
      BFrag k0, k1;
      int dch = i * 32 + lh * 8;
      k0.q[0] = *(const u32x4*)(sK + l15 * 136 + dch);
      k0.q[1] = *(const u32x4*)(sK + l15 * 136 + dch + 16);
      k1.q[0] = *(const u32x4*)(sK + (16 + l15) * 136 + dch);
      k1.q[1] = *(const u32x4*)(sK + (16 + l15) * 136 + dch + 16);
      s0 = wmma_bf16(qf[i].v, k0.v, s0);
      s1 = wmma_bf16(qf[i].v, k1.v, s1);
    }

    // Causal mask + online softmax (rows live across 16-lane halves)
    float p0[8], p1[8], alpha[8];
#pragma unroll
    for (int r = 0; r < 8; ++r) {
      int mrow = qw + r + 8 * lh;
      float x0 = s0[r] * ATT_SCALE;
      float x1 = s1[r] * ATT_SCALE;
      if (j0 + l15 > mrow)      x0 = -3.0e38f;
      if (j0 + 16 + l15 > mrow) x1 = -3.0e38f;
      float mx = fmaxf(x0, x1);
#pragma unroll
      for (int m = 8; m >= 1; m >>= 1) mx = fmaxf(mx, __shfl_xor(mx, m, 16));
      float mnew = fmaxf(mst[r], mx);
      float e0 = __expf(x0 - mnew), e1 = __expf(x1 - mnew);
      float rs = e0 + e1;
#pragma unroll
      for (int m = 8; m >= 1; m >>= 1) rs += __shfl_xor(rs, m, 16);
      alpha[r] = __expf(mst[r] - mnew);
      lst[r]   = lst[r] * alpha[r] + rs;
      mst[r]   = mnew;
      p0[r] = e0; p1[r] = e1;
    }
#pragma unroll
    for (int c = 0; c < 8; ++c)
#pragma unroll
      for (int r = 0; r < 8; ++r) o[c][r] *= alpha[r];

    // Relayout P (C layout -> A layout) through per-wave LDS scratch
    bf16* pw = sP + w * 16 * LDT;
#pragma unroll
    for (int r = 0; r < 8; ++r) {
      int pr = r + 8 * lh;
      pw[pr * LDT + l15]      = f2bf(p0[r]);
      pw[pr * LDT + 16 + l15] = f2bf(p1[r]);
    }
    asm volatile("s_wait_dscnt 0" ::: "memory");  // wave-local LDS RAW fence
    AFrag pf;
    pf.q[0] = *(const u32x4*)(pw + l15 * LDT + lh * 8);
    pf.q[1] = *(const u32x4*)(pw + l15 * LDT + lh * 8 + 16);

    // O += P @ V  (V^T in LDS gives contiguous B fragments)
#pragma unroll
    for (int c = 0; c < 8; ++c) {
      BFrag vf;
      int dr = c * 16 + l15;
      vf.q[0] = *(const u32x4*)(sVt + dr * LDT + lh * 8);
      vf.q[1] = *(const u32x4*)(sVt + dr * LDT + lh * 8 + 16);
      o[c] = wmma_bf16(pf.v, vf.v, o[c]);
    }
  }

  // Normalize and store (bf16, [B*T, H] with head column offset)
  float inv[8];
#pragma unroll
  for (int r = 0; r < 8; ++r) inv[r] = 1.0f / lst[r];
#pragma unroll
  for (int c = 0; c < 8; ++c)
#pragma unroll
    for (int r = 0; r < 8; ++r) {
      int m = qw + r + 8 * lh;
      Ob[base + (size_t)m * H + c * 16 + l15] = f2bf(o[c][r] * inv[r]);
    }
}

// ---------------------------------------------------------------------------
// Launch
// ---------------------------------------------------------------------------
extern "C" void kernel_launch(void* const* d_in, const int* in_sizes, int n_in,
                              void* d_out, int out_size, void* d_ws, size_t ws_size,
                              hipStream_t stream) {
  (void)in_sizes; (void)n_in; (void)out_size; (void)ws_size;
  const float* x     = (const float*)d_in[0];
  const float* gamma = (const float*)d_in[1];
  const float* beta  = (const float*)d_in[2];
  const float* Wq    = (const float*)d_in[3];
  const float* Wk    = (const float*)d_in[4];
  const float* Wv    = (const float*)d_in[5];
  const float* Wo    = (const float*)d_in[6];
  float* out = (float*)d_out;

  char* ws = (char*)d_ws;
  const size_t MB = 1ull << 20;
  bf16* xn  = (bf16*)(ws + 0 * MB);
  bf16* Qb  = (bf16*)(ws + 16 * MB);
  bf16* Kb  = (bf16*)(ws + 32 * MB);
  bf16* Vb  = (bf16*)(ws + 48 * MB);
  bf16* Ob  = (bf16*)(ws + 64 * MB);
  bf16* WqB = (bf16*)(ws + 80 * MB);
  bf16* WkB = (bf16*)(ws + 82 * MB);
  bf16* WvB = (bf16*)(ws + 84 * MB);
  bf16* WoB = (bf16*)(ws + 86 * MB);

  // 1) weights fp32 -> bf16 (1M elements each)
  cvt_w_kernel<<<dim3((H * H) / 256, 4), 256, 0, stream>>>(
      Wq, Wk, Wv, Wo, WqB, WkB, WvB, WoB);
  // 2) LayerNorm -> bf16
  ln_kernel<<<NB * T, 256, 0, stream>>>(x, gamma, beta, xn);
  // 3) Q/K/V projections (WMMA + TDM staging)
  gemm_qkv_kernel<<<dim3(H / 64, (NB * T) / 128, 3), 256, 0, stream>>>(
      xn, WqB, WkB, WvB, Qb, Kb, Vb);
  // 4) causal flash attention (WMMA + TDM K staging)
  attn_kernel<<<dim3(T / 64, NH, NB), 128, 0, stream>>>(Qb, Kb, Vb, Ob);
  // 5) output projection + residual (WMMA, fp32 out)
  gemm_resid_kernel<<<dim3(H / 64, (NB * T) / 128), 256, 0, stream>>>(
      Ob, WoB, x, out);
}
